// GRUModel_29265907155193
// MI455X (gfx1250) — compile-verified, matrix operand order checked by reading, not run on hardware
//
#include <hip/hip_runtime.h>
#include <hip/hip_bf16.h>

typedef __attribute__((ext_vector_type(16))) _Float16 v16h;
typedef __attribute__((ext_vector_type(8)))  _Float16 v8h;
typedef __attribute__((ext_vector_type(8)))  float    v8f;

#define T_LEN 512
#define HID   64
#define G3H   192
#define MROWS 16
#define BLOCK 128
#define XCHUNK 64
#define HSZ   (MROWS * HID)

#define LOG2E  1.4426950408889634f

// sigmoid(v) = 1 / (1 + 2^(-log2e * v)); v->-inf: exp2->inf, rcp(inf)=0  (safe)
__device__ __forceinline__ float sigm(float v) {
  return __builtin_amdgcn_rcpf(1.0f + __builtin_amdgcn_exp2f(-LOG2E * v));
}
// tanh(v) = 2 / (1 + 2^(-2*log2e*v)) - 1; saturates to +-1 at both extremes
__device__ __forceinline__ float tanh_fast(float v) {
#if __has_builtin(__builtin_amdgcn_tanhf)
  return __builtin_amdgcn_tanhf(v);          // native V_TANH_F32 on gfx1250
#else
  return 2.0f * __builtin_amdgcn_rcpf(1.0f + __builtin_amdgcn_exp2f(-2.0f * LOG2E * v)) - 1.0f;
#endif
}

// WMMA A/B fragment from an f16 [row][64] array (global-converted or LDS).
// Per ISA 16-bit layout: lanes 0-15 -> K base 0, lanes 16-31 -> K base 8;
// elements 0..7 hold K=kb..kb+7, elements 8..15 hold K=kb+16..kb+23.
// Both 8-half runs are contiguous -> two b128 loads.
__device__ __forceinline__ v16h load_frag(const _Float16* base, int row,
                                          int kchunk, int hi16) {
  int kb = 32 * kchunk + (hi16 ? 8 : 0);
  v8h lo = *(const v8h*)(base + row * HID + kb);
  v8h hi = *(const v8h*)(base + row * HID + kb + 16);
  return __builtin_shufflevector(lo, hi, 0, 1, 2, 3, 4, 5, 6, 7,
                                 8, 9, 10, 11, 12, 13, 14, 15);
}

// Same fragment gathered from a global f32 weight matrix (startup only).
__device__ __forceinline__ v16h load_bfrag_f32(const float* __restrict__ w,
                                               int n, int kchunk, int hi16) {
  v16h f;
  int kb = 32 * kchunk + (hi16 ? 8 : 0);
#pragma unroll
  for (int e = 0; e < 8; ++e) {
    f[e]     = (_Float16)w[n * HID + kb + e];
    f[e + 8] = (_Float16)w[n * HID + kb + 16 + e];
  }
  return f;
}

__device__ __forceinline__ v8f splat8(float b) {
  v8f c = {b, b, b, b, b, b, b, b};
  return c;
}

#define WMMA(a, b, c) \
  __builtin_amdgcn_wmma_f32_16x16x32_f16(false, (a), false, (b), (short)0, (c), false, false)

__global__ __launch_bounds__(BLOCK) void gru2_fused(
    const float* __restrict__ x,
    const float* __restrict__ w_ih0, const float* __restrict__ w_hh0,
    const float* __restrict__ b_ih0, const float* __restrict__ b_hh0,
    const float* __restrict__ w_ih1, const float* __restrict__ w_hh1,
    const float* __restrict__ b_ih1, const float* __restrict__ b_hh1,
    const float* __restrict__ fc_w,  const float* __restrict__ fc_b,
    float* __restrict__ out) {
  __shared__ _Float16 w1i_lds[G3H * HID];      // f16 w_ih1, [n][k]   (24 KB)
  __shared__ _Float16 w1h_lds[G3H * HID];      // f16 w_hh1, [n][k]   (24 KB)
  __shared__ _Float16 h1_lds[2][HSZ];          // ping-pong layer-1 h (4 KB)
  __shared__ _Float16 h2_lds[2][HSZ];          // ping-pong layer-2 h (4 KB)
  __shared__ float    x_lds[XCHUNK * MROWS];   // x chunk, [t][row]   (4 KB)

  const int tid  = threadIdx.x;
  const int wv   = tid >> 5;   // wave 0..3: owns hidden cols 16*wv..16*wv+15
  const int lane = tid & 31;
  const int lh   = lane & 15;
  const int hi16 = lane >> 4;
  const int b0   = blockIdx.x * MROWS;

  // Stage layer-1 weights to LDS as f16 (read every step, written never again).
  for (int idx = tid; idx < G3H * HID; idx += BLOCK) {
    w1i_lds[idx] = (_Float16)w_ih1[idx];
    w1h_lds[idx] = (_Float16)w_hh1[idx];
  }
  for (int idx = tid; idx < HSZ; idx += BLOCK) {
    h1_lds[0][idx] = (_Float16)0.0f;
    h2_lds[0][idx] = (_Float16)0.0f;
  }

  // Layer-0 recurrent weights stay register-resident (6 x v16h = 48 VGPRs).
  v16h Bhh0[3][2];
  int   nrow[3];
  float wi0[3], bi0[3], bh0[3], bi1[3], bh1[3];
#pragma unroll
  for (int c = 0; c < 3; ++c) {
    int n = 64 * c + 16 * wv + lh;
    nrow[c] = n;
    wi0[c] = w_ih0[n];  // Din == 1
    bi0[c] = b_ih0[n];
    bh0[c] = b_hh0[n];
    bi1[c] = b_ih1[n];
    bh1[c] = b_hh1[n];
    Bhh0[c][0] = load_bfrag_f32(w_hh0, n, 0, hi16);
    Bhh0[c][1] = load_bfrag_f32(w_hh0, n, 1, hi16);
  }

  // f32 master copies of this wave's hidden-state slice (D-layout).
  v8f h1s = splat8(0.0f), h2s = splat8(0.0f);

  int zoff = 0;  // opaque zero: defeats hoisting of LDS weight loads
  __syncthreads();

  for (int t = 0; t < T_LEN; ++t) {
    const int p = t & 1, q = p ^ 1;

    // Refill x chunk every XCHUNK steps (prev step's barrier covers old reads).
    if ((t & (XCHUNK - 1)) == 0) {
      for (int idx = tid; idx < XCHUNK * MROWS; idx += BLOCK) {
        int tt = idx >> 4, row = idx & 15;
        x_lds[idx] = x[(size_t)(b0 + row) * T_LEN + t + tt];
      }
      __syncthreads();
    }

    asm volatile("" : "+v"(zoff));  // force per-iteration LDS weight reloads
    const _Float16* w1i = w1i_lds + zoff;
    const _Float16* w1h = w1h_lds + zoff;

    // ---- layer 0: gh0 = h1_old @ w_hh0^T + b_hh0 (write into alt buffer) ----
    {
      v16h a0 = load_frag(h1_lds[p], lh, 0, hi16);
      v16h a1 = load_frag(h1_lds[p], lh, 1, hi16);
      v8f accA[3];
#pragma unroll
      for (int c = 0; c < 3; ++c) {
        accA[c] = splat8(bh0[c]);
        accA[c] = WMMA(a0, Bhh0[c][0], accA[c]);
        accA[c] = WMMA(a1, Bhh0[c][1], accA[c]);
      }
#pragma unroll
      for (int e = 0; e < 8; ++e) {
        float xv  = x_lds[(t & (XCHUNK - 1)) * MROWS + (e + 8 * hi16)];
        float r = sigm(xv * wi0[0] + bi0[0] + accA[0][e]);
        float z = sigm(xv * wi0[1] + bi0[1] + accA[1][e]);
        float n = tanh_fast(xv * wi0[2] + bi0[2] + r * accA[2][e]);
        h1s[e] = (1.0f - z) * n + z * h1s[e];
      }
#pragma unroll
      for (int e = 0; e < 8; ++e)
        h1_lds[q][(e + 8 * hi16) * HID + 16 * wv + lh] = (_Float16)h1s[e];
    }
    __syncthreads();  // h1_new visible to all waves

    // ---- layer 1: gi1 = h1_new @ w_ih1^T ; gh1 = h2_old @ w_hh1^T ----
    {
      v16h a1n0 = load_frag(h1_lds[q], lh, 0, hi16);
      v16h a1n1 = load_frag(h1_lds[q], lh, 1, hi16);
      v8f accI[3];
#pragma unroll
      for (int c = 0; c < 3; ++c) {
        accI[c] = splat8(bi1[c]);
        accI[c] = WMMA(a1n0, load_frag(w1i, nrow[c], 0, hi16), accI[c]);
        accI[c] = WMMA(a1n1, load_frag(w1i, nrow[c], 1, hi16), accI[c]);
      }
      v16h a2o0 = load_frag(h2_lds[p], lh, 0, hi16);
      v16h a2o1 = load_frag(h2_lds[p], lh, 1, hi16);
      v8f accH[3];
#pragma unroll
      for (int c = 0; c < 3; ++c) {
        accH[c] = splat8(bh1[c]);
        accH[c] = WMMA(a2o0, load_frag(w1h, nrow[c], 0, hi16), accH[c]);
        accH[c] = WMMA(a2o1, load_frag(w1h, nrow[c], 1, hi16), accH[c]);
      }
#pragma unroll
      for (int e = 0; e < 8; ++e) {
        float r = sigm(accI[0][e] + accH[0][e]);
        float z = sigm(accI[1][e] + accH[1][e]);
        float n = tanh_fast(accI[2][e] + r * accH[2][e]);
        h2s[e] = (1.0f - z) * n + z * h2s[e];
      }
#pragma unroll
      for (int e = 0; e < 8; ++e)
        h2_lds[q][(e + 8 * hi16) * HID + 16 * wv + lh] = (_Float16)h2s[e];
    }
    __syncthreads();  // h1[q]/h2[q] ready for next step's reads
  }

  // ---- epilogue: out[b] = h2_last[b] . fc_w + fc_b (f32, reuse x_lds) ----
  float* h2f = x_lds;  // 1024 floats == 16x64, exact fit
#pragma unroll
  for (int e = 0; e < 8; ++e)
    h2f[(e + 8 * hi16) * HID + 16 * wv + lh] = h2s[e];
  __syncthreads();
  if (tid < MROWS) {
    float acc = fc_b[0];
#pragma unroll 8
    for (int k = 0; k < HID; ++k) acc += h2f[tid * HID + k] * fc_w[k];
    out[b0 + tid] = acc;
  }
}

extern "C" void kernel_launch(void* const* d_in, const int* in_sizes, int n_in,
                              void* d_out, int out_size, void* d_ws, size_t ws_size,
                              hipStream_t stream) {
  const float* x     = (const float*)d_in[0];
  const float* w_ih0 = (const float*)d_in[1];
  const float* w_hh0 = (const float*)d_in[2];
  const float* b_ih0 = (const float*)d_in[3];
  const float* b_hh0 = (const float*)d_in[4];
  const float* w_ih1 = (const float*)d_in[5];
  const float* w_hh1 = (const float*)d_in[6];
  const float* b_ih1 = (const float*)d_in[7];
  const float* b_hh1 = (const float*)d_in[8];
  const float* fc_w  = (const float*)d_in[9];
  const float* fc_b  = (const float*)d_in[10];
  float* out = (float*)d_out;

  int B = in_sizes[0] / T_LEN;  // x is [B, T, 1]
  gru2_fused<<<B / MROWS, BLOCK, 0, stream>>>(x, w_ih0, w_hh0, b_ih0, b_hh0,
                                              w_ih1, w_hh1, b_ih1, b_hh1,
                                              fc_w, fc_b, out);
}
